// Spatio_Attention_31464930410843
// MI455X (gfx1250) — matrix operand order repaired and checked by reading, not compiled, not run on hardware
//
#include <hip/hip_runtime.h>
#include <hip/hip_bf16.h>
#include <math.h>

typedef __attribute__((ext_vector_type(16))) _Float16 v16h;
typedef __attribute__((ext_vector_type(8)))  float    v8f;

#define BATCH 2
#define DIM   192
#define HEADS 6
#define CH    32
#define NTOK  2304      // 48*48
#define NT    144       // NTOK/16
#define EPSN  1e-12f

// ---------------------------------------------------------------------------
// Kernel 1: q/k/v projection.  Y(192x2304) = W(192x192) * X(192x2304) per
// batch per matrix, via V_WMMA_F32_16X16X32_F16 (K-loop of 6 chunks of 32).
// Workgroup = 8 waves handles 16 output rows x all 2304 cols.
// ---------------------------------------------------------------------------
__global__ void proj_kernel(const float* __restrict__ x,
                            const float* __restrict__ wq,
                            const float* __restrict__ wk,
                            const float* __restrict__ wv,
                            float* __restrict__ qraw,
                            float* __restrict__ kraw,
                            float* __restrict__ vout)
{
    const int wid  = threadIdx.x >> 5;
    const int lane = threadIdx.x & 31;
    const int half = lane >> 4;
    const int mr   = lane & 15;
    const int nn   = lane & 15;

    int blk   = blockIdx.x;          // b*3*12 + mat*12 + otile
    int otile = blk % 12;  blk /= 12;
    int mat   = blk % 3;
    int b     = blk / 3;

    const float* w = (mat == 0) ? wq : ((mat == 1) ? wk : wv);
    float*       y = (mat == 0) ? qraw : ((mat == 1) ? kraw : vout);
    const int obase = otile * 16;

    // A fragments: A_j[m][kk] = W[obase+m][j*32+kk]  (ISA 16-bit A layout)
    v16h A[6];
#pragma unroll
    for (int j = 0; j < 6; ++j) {
        const float* wr = w + (obase + mr) * DIM + j * 32;
#pragma unroll
        for (int v = 0; v < 8; ++v) {
            int k0 = (v < 4) ? (half * 8 + 2 * v) : (16 + half * 8 + 2 * (v - 4));
            A[j][2 * v]     = (_Float16)wr[k0];
            A[j][2 * v + 1] = (_Float16)wr[k0 + 1];
        }
    }

    const float* xb = x + (size_t)b * DIM * NTOK;
    for (int t = wid; t < NT; t += 8) {
        const int colbase = t * 16;
        v8f c = {};
#pragma unroll
        for (int j = 0; j < 6; ++j) {
            // B[kk][nn] = X[j*32+kk][colbase+nn]  (ISA 16-bit B layout)
            v16h Bm;
            const float* xc = xb + colbase + nn;
#pragma unroll
            for (int v = 0; v < 8; ++v) {
                int k0 = (v < 4) ? (half * 8 + 2 * v) : (16 + half * 8 + 2 * (v - 4));
                Bm[2 * v]     = (_Float16)xc[(size_t)(j * 32 + k0) * NTOK];
                Bm[2 * v + 1] = (_Float16)xc[(size_t)(j * 32 + k0 + 1) * NTOK];
            }
            c = __builtin_amdgcn_wmma_f32_16x16x32_f16(false, A[j], false, Bm,
                                                       (short)0, c, false, false);
        }
        // D layout: lanes0-15 -> M=r, lanes16-31 -> M=8+r, N=lane&15
        float* yb = y + (size_t)b * DIM * NTOK + (size_t)(obase + half * 8) * NTOK
                      + colbase + nn;
#pragma unroll
        for (int r = 0; r < 8; ++r) yb[(size_t)r * NTOK] = c[r];
    }
}

// ---------------------------------------------------------------------------
// Kernel 2: L2-normalize channel pairs of q and k over n (2304) and emit a
// packed f16 token-major fragment layout:
//   pack[bh][n][p]  (p = 0..15 dwords) = f16pair(chan 2p, chan 2p+1) of token n
// so the attention kernel can fetch A/B fragments with 2x global_load_b128.
// Temperature[head] is folded into the q-side scale.
// One workgroup per (mat, b, channel-pair).
// ---------------------------------------------------------------------------
__global__ void norm_pack_kernel(const float* __restrict__ qraw,
                                 const float* __restrict__ kraw,
                                 const float* __restrict__ temp,
                                 unsigned* __restrict__ qpack,
                                 unsigned* __restrict__ kpack)
{
    int blk = blockIdx.x;                 // mat(2) * B(2) * 96 pairs
    int p   = blk % 96;  blk /= 96;
    int b   = blk % BATCH;
    int mat = blk / BATCH;                // 0 = q, 1 = k
    const int h  = p >> 4;                // head = pair/16
    const int c0 = 2 * p;

    const float* src = (mat == 0 ? qraw : kraw) + ((size_t)b * DIM + c0) * NTOK;
    const float* r0 = src;
    const float* r1 = src + NTOK;

    __shared__ float red0[256], red1[256];
    float s0 = 0.f, s1 = 0.f;
    for (int j = threadIdx.x; j < NTOK; j += 256) {
        float a = r0[j], bb = r1[j];
        s0 += a * a;  s1 += bb * bb;
    }
    red0[threadIdx.x] = s0;  red1[threadIdx.x] = s1;
    __syncthreads();
    for (int st = 128; st > 0; st >>= 1) {
        if (threadIdx.x < (unsigned)st) {
            red0[threadIdx.x] += red0[threadIdx.x + st];
            red1[threadIdx.x] += red1[threadIdx.x + st];
        }
        __syncthreads();
    }
    float inv0 = 1.0f / fmaxf(sqrtf(red0[0]), EPSN);
    float inv1 = 1.0f / fmaxf(sqrtf(red1[0]), EPSN);
    if (mat == 0) { float tp = temp[h]; inv0 *= tp; inv1 *= tp; }

    unsigned* dst = (mat == 0 ? qpack : kpack)
                  + ((size_t)(b * HEADS + h) * NTOK) * 16 + (p & 15);
    for (int j = threadIdx.x; j < NTOK; j += 256) {
        union { _Float16 hh[2]; unsigned u; } pd;
        pd.hh[0] = (_Float16)(r0[j] * inv0);
        pd.hh[1] = (_Float16)(r1[j] * inv1);
        dst[(size_t)j * 16] = pd.u;
    }
}

// ---------------------------------------------------------------------------
// Kernel 3: fused attn = softmax(k_hat^T q_hat).  One workgroup per
// (b, head, 16-row strip).  8 waves build the 16x2304 strip in LDS via one
// WMMA per 16x16 tile (K = ch = 32 in a single instruction); A/B fragments
// come from the packed layout as 2x b128 loads each.  Then row softmax in
// LDS and a single coalesced HBM write of the final attn.
// ---------------------------------------------------------------------------
__global__ void attn_kernel(const unsigned* __restrict__ kpack,
                            const unsigned* __restrict__ qpack,
                            float* __restrict__ out)
{
    extern __shared__ float strip[];      // [16][NTOK] = 147456 bytes
    const int wid  = threadIdx.x >> 5;
    const int lane = threadIdx.x & 31;
    const int half = lane >> 4;
    const int mr   = lane & 15;
    const int nn   = lane & 15;

    const int rowTile = blockIdx.x % NT;
    const int bh      = blockIdx.x / NT;  // 0..11 = b*HEADS + h
    const int rowbase = rowTile * 16;

    const uint4* kp = (const uint4*)kpack + (size_t)bh * NTOK * 4;
    const uint4* qp = (const uint4*)qpack + (size_t)bh * NTOK * 4;

    union FragU { uint4 q[2]; v16h h; };

    // A fragment: token row n = rowbase+mr; pair-dwords [half*4..+3], [8+half*4..+3]
    FragU af;
    {
        const int na = rowbase + mr;
        af.q[0] = kp[na * 4 + half];
        af.q[1] = kp[na * 4 + 2 + half];
    }
    const v16h A = af.h;

    for (int t = wid; t < NT; t += 8) {
        // B fragment: token col n = t*16+nn; pair-dwords [half*4..+3], [8+half*4..+3]
        const int nb = t * 16 + nn;
        FragU bf;
        bf.q[0] = qp[nb * 4 + half];
        bf.q[1] = qp[nb * 4 + 2 + half];

        v8f c = {};
        c = __builtin_amdgcn_wmma_f32_16x16x32_f16(false, A, false, bf.h,
                                                   (short)0, c, false, false);
        float* dst = strip + (size_t)(half * 8) * NTOK + t * 16 + nn;
#pragma unroll
        for (int r = 0; r < 8; ++r) dst[(size_t)r * NTOK] = c[r];
    }
    __syncthreads();

    // Softmax: each wave owns 2 rows of the strip (8 waves * 2 = 16 rows).
#pragma unroll
    for (int rr = 0; rr < 2; ++rr) {
        const int row = wid * 2 + rr;
        float* srow = strip + (size_t)row * NTOK;

        float m = -3.402823e38f;
        for (int j = lane; j < NTOK; j += 32) m = fmaxf(m, srow[j]);
#pragma unroll
        for (int off = 16; off > 0; off >>= 1) m = fmaxf(m, __shfl_xor(m, off, 32));

        float s = 0.f;
        for (int j = lane; j < NTOK; j += 32) {
            float e = __expf(srow[j] - m);
            srow[j] = e;
            s += e;
        }
#pragma unroll
        for (int off = 16; off > 0; off >>= 1) s += __shfl_xor(s, off, 32);

        const float inv = 1.0f / s;
        float* orow = out + ((size_t)bh * NTOK + rowbase + row) * (size_t)NTOK;
        for (int j = lane; j < NTOK; j += 32) orow[j] = srow[j] * inv;
    }
}

// ---------------------------------------------------------------------------
extern "C" void kernel_launch(void* const* d_in, const int* in_sizes, int n_in,
                              void* d_out, int out_size, void* d_ws, size_t ws_size,
                              hipStream_t stream)
{
    const float* inp  = (const float*)d_in[0];
    const float* wq   = (const float*)d_in[1];
    const float* wk   = (const float*)d_in[2];
    const float* wv   = (const float*)d_in[3];
    const float* temp = (const float*)d_in[4];

    float* out = (float*)d_out;
    const size_t attnElems = (size_t)BATCH * HEADS * NTOK * NTOK;  // 63,700,992
    float* vout = out + attnElems;                                 // v region

    // workspace layout (16B-aligned chunks):
    //   qraw  : [B][DIM][NTOK] f32   (3.54 MB)
    //   kraw  : [B][DIM][NTOK] f32   (3.54 MB)
    //   qpack : [B*HEADS][NTOK][16] u32 (1.77 MB)  f16 channel-pair packed
    //   kpack : [B*HEADS][NTOK][16] u32 (1.77 MB)
    float*    qraw  = (float*)d_ws;
    float*    kraw  = qraw + (size_t)BATCH * DIM * NTOK;
    unsigned* qpack = (unsigned*)(kraw + (size_t)BATCH * DIM * NTOK);
    unsigned* kpack = qpack + (size_t)BATCH * HEADS * NTOK * 16;

    // 1) projections: 2 batches * 3 mats * 12 row-tiles workgroups
    proj_kernel<<<dim3(BATCH * 3 * 12), dim3(256), 0, stream>>>(
        inp, wq, wk, wv, qraw, kraw, vout);

    // 2) row L2-normalization (+ temperature folded into q) + f16 fragment pack
    norm_pack_kernel<<<dim3(2 * BATCH * 96), dim3(256), 0, stream>>>(
        qraw, kraw, temp, qpack, kpack);

    // 3) fused attention + softmax; 147456 B dynamic LDS strip per workgroup
    attn_kernel<<<dim3(BATCH * HEADS * NT), dim3(256), 16 * NTOK * 4, stream>>>(
        kpack, qpack, out);
}
